// DGCNN_Seg_5609227289173
// MI455X (gfx1250) — matrix operand order repaired and checked by reading, not compiled, not run on hardware
//
#include <hip/hip_runtime.h>

typedef __attribute__((ext_vector_type(16))) _Float16     v16h;
typedef __attribute__((ext_vector_type(8)))  float        v8f;
typedef __attribute__((ext_vector_type(4)))  unsigned int v4u;

union Frag { v4u u[2]; v16h h; };

#define RTOT 16384   // B*N
#define NPTS 4096    // N per batch
#define TOPK 20
#define EPS  1e-5f

// ---------------------------------------------------------------------------
// Fragment loaders (16x16x32 f16 WMMA layouts, wave32)
// A (16x32, MxK): lane l: m=l&15, holds K = g8..g8+7 and 16+g8..16+g8+7, g8=(l>>4)*8
// B (32x16, KxN) from row-major Bt[n][k]: lane l: n=l&15, holds K = g8..g8+7 and 16+g8..16+g8+7, g8=g16/2
// ---------------------------------------------------------------------------
__device__ __forceinline__ v16h frag_a(const _Float16* rowPtr, int g8) {
  Frag f;
  f.u[0] = *(const v4u*)(rowPtr + g8);
  f.u[1] = *(const v4u*)(rowPtr + g8 + 16);
  return f.h;
}
__device__ __forceinline__ v16h frag_b(const _Float16* rowPtr, int g16) {
  Frag f;
  f.u[0] = *(const v4u*)(rowPtr + g16 / 2);
  f.u[1] = *(const v4u*)(rowPtr + g16 / 2 + 16);
  return f.h;
}

// ---------------------------------------------------------------------------
// Epilogue store for one 16x16 accumulator tile: optional per-batch row-bias,
// BN (y*s+t), col-bias, LeakyReLU; stores f32 and/or f16.
// ---------------------------------------------------------------------------
__device__ __forceinline__ void gemm_store(
    v8f acc, int r0, int c0, int lane,
    float* __restrict__ o32, _Float16* __restrict__ o16, int ldo, int ncols_store,
    const float* __restrict__ s, const float* __restrict__ t,
    const float* __restrict__ rowbias, int rb_ld, int n_per_batch,
    const float* __restrict__ colbias, int lrelu)
{
  int col = c0 + (lane & 15);
  if (col >= ncols_store) return;
  float sc = s ? s[col] : 1.0f;
  float sh = t ? t[col] : 0.0f;
  float cb = colbias ? colbias[col] : 0.0f;
#pragma unroll
  for (int v = 0; v < 8; ++v) {
    int r = r0 + ((lane >> 4) * 8) + v;          // C layout: M=(l>=16)*8+v, N=l&15
    float y = acc[v];
    if (rowbias) y += rowbias[(r / n_per_batch) * rb_ld + col];
    y = y * sc + sh + cb;
    if (lrelu) y = (y >= 0.0f) ? y : 0.2f * y;
    if (o32) o32[(size_t)r * ldo + col] = y;
    if (o16) o16[(size_t)r * ldo + col] = (_Float16)y;
  }
}

// ---------------------------------------------------------------------------
// Generic WMMA GEMM: Y[r][c] = sum_k A[r][k] * Bt[c][k].
// One wave computes a 32x32 output block (2x2 WMMA tiles): each A/B fragment
// is reused across two WMMAs -> 128 B loaded per WMMA (vs 256 for 1x1).
// ---------------------------------------------------------------------------
__global__ __launch_bounds__(256) void gemm_wmma(
    const _Float16* __restrict__ A, int lda,
    const _Float16* __restrict__ Bt, int ldb,
    float* __restrict__ o32, _Float16* __restrict__ o16, int ldo,
    int tiles_n, int n_tiles_total, int kdim, int ncols_store,
    const float* __restrict__ s, const float* __restrict__ t,
    const float* __restrict__ rowbias, int rb_ld, int n_per_batch,
    const float* __restrict__ colbias, int lrelu)
{
  int wid = blockIdx.x * (blockDim.x >> 5) + (threadIdx.x >> 5);
  if (wid >= n_tiles_total) return;                 // wave-uniform: EXEC stays full
  int tm = wid / tiles_n, tn = wid % tiles_n;
  int row0 = tm * 32, col0 = tn * 32;
  int lane = threadIdx.x & 31;
  int m   = lane & 15;
  int g8  = (lane >> 4) * 8;
  int g16 = (lane >> 4) * 16;

  const _Float16* aR0 = A  + (size_t)(row0 + m) * lda;
  const _Float16* aR1 = aR0 + (size_t)16 * lda;
  const _Float16* bR0 = Bt + (size_t)(col0 + m) * ldb;
  const _Float16* bR1 = bR0 + (size_t)16 * ldb;

  v8f acc00 = {}, acc01 = {}, acc10 = {}, acc11 = {};
  for (int k0 = 0; k0 < kdim; k0 += 32) {
    v16h a0 = frag_a(aR0 + k0, g8);
    v16h a1 = frag_a(aR1 + k0, g8);
    v16h b0 = frag_b(bR0 + k0, g16);
    v16h b1 = frag_b(bR1 + k0, g16);
    acc00 = __builtin_amdgcn_wmma_f32_16x16x32_f16(false, a0, false, b0,
                                                   (short)0, acc00, false, false);
    acc01 = __builtin_amdgcn_wmma_f32_16x16x32_f16(false, a0, false, b1,
                                                   (short)0, acc01, false, false);
    acc10 = __builtin_amdgcn_wmma_f32_16x16x32_f16(false, a1, false, b0,
                                                   (short)0, acc10, false, false);
    acc11 = __builtin_amdgcn_wmma_f32_16x16x32_f16(false, a1, false, b1,
                                                   (short)0, acc11, false, false);
  }

  gemm_store(acc00, row0,      col0,      lane, o32, o16, ldo, ncols_store,
             s, t, rowbias, rb_ld, n_per_batch, colbias, lrelu);
  gemm_store(acc01, row0,      col0 + 16, lane, o32, o16, ldo, ncols_store,
             s, t, rowbias, rb_ld, n_per_batch, colbias, lrelu);
  gemm_store(acc10, row0 + 16, col0,      lane, o32, o16, ldo, ncols_store,
             s, t, rowbias, rb_ld, n_per_batch, colbias, lrelu);
  gemm_store(acc11, row0 + 16, col0 + 16, lane, o32, o16, ldo, ncols_store,
             s, t, rowbias, rb_ld, n_per_batch, colbias, lrelu);
}

// ---------------------------------------------------------------------------
// kNN: one wave per 16 query rows. Gram tile via WMMA, distances through LDS,
// per-row top-20 sorted lists in registers (lanes 0..15, one row each).
// Writes absolute (batch-global) neighbor row indices.
// ---------------------------------------------------------------------------
__global__ __launch_bounds__(32) void knn_wmma(
    const _Float16* __restrict__ feat, int ldf, int cdim,
    const float* __restrict__ xx, int* __restrict__ idx, int npts)
{
  __shared__ float sg[256];
  __shared__ float sxxj[16];
  int lane  = threadIdx.x;
  int i0    = blockIdx.x * 16;
  int jbase = (i0 / npts) * npts;
  int m   = lane & 15;
  int g8  = (lane >> 4) * 8;
  int g16 = (lane >> 4) * 16;

  float td[TOPK]; int ti[TOPK];
#pragma unroll
  for (int q = 0; q < TOPK; ++q) { td[q] = 3.0e38f; ti[q] = jbase; }
  float xi = (lane < 16) ? xx[i0 + lane] : 0.0f;

  const _Float16* aRow = feat + (size_t)(i0 + m) * ldf;

  for (int j0 = 0; j0 < npts; j0 += 16) {
    const _Float16* bRow = feat + (size_t)(jbase + j0 + m) * ldf;
    v8f acc = {};
    for (int k0 = 0; k0 < cdim; k0 += 32) {
      v16h a = frag_a(aRow + k0, g8);
      v16h b = frag_b(bRow + k0, g16);
      acc = __builtin_amdgcn_wmma_f32_16x16x32_f16(false, a, false, b,
                                                   (short)0, acc, false, false);
    }
    if (lane < 16) sxxj[lane] = xx[jbase + j0 + lane];
#pragma unroll
    for (int v = 0; v < 8; ++v)
      sg[((lane >> 4) * 8 + v) * 16 + m] = acc[v];   // C layout: M=(l>=16)*8+v, N=l&15
    __syncthreads();

    if (lane < 16) {
      for (int n = 0; n < 16; ++n) {
        float d = xi + sxxj[n] - 2.0f * sg[lane * 16 + n];
        if (d < td[TOPK - 1]) {
          td[TOPK - 1] = d; ti[TOPK - 1] = jbase + j0 + n;
#pragma unroll
          for (int q = TOPK - 1; q > 0; --q) {       // single bubble pass inserts
            if (td[q] < td[q - 1]) {
              float fd = td[q]; td[q] = td[q - 1]; td[q - 1] = fd;
              int   fi = ti[q]; ti[q] = ti[q - 1]; ti[q - 1] = fi;
            }
          }
        }
      }
    }
    __syncthreads();
  }
  if (lane < 16) {
#pragma unroll
    for (int q = 0; q < TOPK; ++q)
      idx[(size_t)(i0 + lane) * TOPK + q] = ti[q];
  }
}

// ---------------------------------------------------------------------------
// EdgeConv epilogue: out = lrelu(bn(max_j V[idx_j] + U)), written f16 into the
// concatenated feature buffer at column offset. One block per point.
// ---------------------------------------------------------------------------
__global__ void gather_max_bn(const float* __restrict__ UV, int lduv, int cout,
                              const int* __restrict__ idx,
                              const float* __restrict__ s, const float* __restrict__ t,
                              _Float16* __restrict__ out, int ldo, int coff)
{
  __shared__ int sidx[TOPK];
  int r = blockIdx.x, c = threadIdx.x;
  if (c < TOPK) sidx[c] = idx[(size_t)r * TOPK + c];
  __syncthreads();
  float mv = -3.0e38f;
#pragma unroll
  for (int j = 0; j < TOPK; ++j)
    mv = fmaxf(mv, UV[(size_t)sidx[j] * lduv + c]);
  float y = mv + UV[(size_t)r * lduv + cout + c];
  y = y * s[c] + t[c];
  y = (y >= 0.0f) ? y : 0.2f * y;
  out[(size_t)r * ldo + coff + c] = (_Float16)y;
}

// ---------------------------------------------------------------------------
// Row squared norms (f16 feats -> f32), one wave per row, wave32 shfl reduce.
// ---------------------------------------------------------------------------
__global__ __launch_bounds__(128) void row_norms(
    const _Float16* __restrict__ feat, int ldf, int cdim,
    float* __restrict__ xx, int rows)
{
  int w = threadIdx.x >> 5, lane = threadIdx.x & 31;
  int r = blockIdx.x * 4 + w;
  if (r >= rows) return;
  float s = 0.0f;
  for (int c = lane; c < cdim; c += 32) {
    float v = (float)feat[(size_t)r * ldf + c];
    s += v * v;
  }
#pragma unroll
  for (int msk = 16; msk; msk >>= 1) s += __shfl_xor(s, msk, 32);
  if (lane == 0) xx[r] = s;
}

// ---------------------------------------------------------------------------
// Small prep / reduction kernels
// ---------------------------------------------------------------------------
__global__ void cvt_xyz(const float* __restrict__ xyz, _Float16* __restrict__ out, int rows)
{
  int id = blockIdx.x * blockDim.x + threadIdx.x;
  if (id >= rows * 32) return;
  int r = id >> 5, c = id & 31;
  out[id] = (_Float16)((c < 3) ? xyz[r * 3 + c] : 0.0f);
}

// W [Cout][2C] -> Wt [2*Cout][Cpad] f16: rows 0..Cout-1 = W1, rows Cout.. = W2-W1
__global__ void prep_edge_w(const float* __restrict__ w, int Cout, int C, int Cpad,
                            _Float16* __restrict__ wt)
{
  int id = blockIdx.x * blockDim.x + threadIdx.x;
  if (id >= 2 * Cout * Cpad) return;
  int o2 = id / Cpad, c = id % Cpad;
  float val = 0.0f;
  if (c < C) {
    if (o2 < Cout) val = w[(size_t)o2 * 2 * C + c];
    else {
      int o = o2 - Cout;
      val = w[(size_t)o * 2 * C + C + c] - w[(size_t)o * 2 * C + c];
    }
  }
  wt[id] = (_Float16)val;
}

// W [O][src_ld] -> Wt [Opad][I] f16 (zero-pad extra rows); copies cols [0,I)
__global__ void prep_plain_w(const float* __restrict__ w, int O, int src_ld, int I,
                             int Opad, _Float16* __restrict__ wt)
{
  int id = blockIdx.x * blockDim.x + threadIdx.x;
  if (id >= Opad * I) return;
  int o = id / I, i = id % I;
  wt[id] = (_Float16)((o < O) ? w[(size_t)o * src_ld + i] : 0.0f);
}

__global__ void prep_bn(const float* __restrict__ g, const float* __restrict__ b,
                        const float* __restrict__ m, const float* __restrict__ v,
                        int n, float* __restrict__ s, float* __restrict__ t)
{
  int i = blockIdx.x * blockDim.x + threadIdx.x;
  if (i >= n) return;
  float sc = g[i] * rsqrtf(v[i] + EPS);
  s[i] = sc;
  t[i] = b[i] - m[i] * sc;
}

// g[b][c] = max_n f[b*npts+n][c]
__global__ void col_max(const _Float16* __restrict__ f, int ld, int npts,
                        float* __restrict__ g)
{
  int b = blockIdx.x, c = blockIdx.y * blockDim.x + threadIdx.x;
  const _Float16* p = f + (size_t)b * npts * ld + c;
  float mv = -3.0e38f;
  for (int n = 0; n < npts; ++n) mv = fmaxf(mv, (float)p[(size_t)n * ld]);
  g[b * ld + c] = mv;
}

// c_b[b][o] = sum_c W_h1[o][512+c] * g[b][c]   (per-batch bias replacing g-concat)
__global__ void compute_cb(const float* __restrict__ w, const float* __restrict__ g,
                           float* __restrict__ cb)
{
  int id = blockIdx.x * blockDim.x + threadIdx.x;
  if (id >= 4 * 512) return;
  int b = id / 512, o = id % 512;
  const float* wr = w + (size_t)o * 1536 + 512;
  const float* gr = g + b * 1024;
  float sum = 0.0f;
  for (int c = 0; c < 1024; ++c) sum += wr[c] * gr[c];
  cb[id] = sum;
}

// ---------------------------------------------------------------------------
// Host orchestration
// ---------------------------------------------------------------------------
static void launch_gemm(hipStream_t st,
                        const _Float16* A, int lda, const _Float16* Bt, int ldb,
                        float* o32, _Float16* o16, int ldo,
                        int rows, int ncols_pad, int kdim, int ncols_store,
                        const float* s, const float* t,
                        const float* rowbias, int rb_ld,
                        const float* colbias, int lrelu)
{
  int tiles_m = rows / 32, tiles_n = ncols_pad / 32;   // 32x32 block per wave
  int nT = tiles_m * tiles_n;
  gemm_wmma<<<(nT + 7) / 8, 256, 0, st>>>(A, lda, Bt, ldb, o32, o16, ldo,
                                          tiles_n, nT, kdim, ncols_store,
                                          s, t, rowbias, rb_ld, NPTS, colbias, lrelu);
}

extern "C" void kernel_launch(void* const* d_in, const int* in_sizes, int n_in,
                              void* d_out, int out_size, void* d_ws, size_t ws_size,
                              hipStream_t stream)
{
  (void)in_sizes; (void)n_in; (void)out_size; (void)ws_size;
  const float* xyz = (const float*)d_in[0];
  // param layout (dict order): ec1..ec4: {w,g,b,m,v}, fuse, h1, h2, h3:{w,bias}
  const int ecBase[4] = {1, 6, 11, 16};
  const float* fuse_w = (const float*)d_in[21];
  const float* h1_w   = (const float*)d_in[26];
  const float* h2_w   = (const float*)d_in[31];
  const float* h3_w   = (const float*)d_in[36];
  const float* h3_b   = (const float*)d_in[37];

  // ---- workspace carve-up ----
  size_t off = 0;
  auto carve = [&](size_t bytes) {
    void* p = (char*)d_ws + off;
    off = (off + bytes + 255) & ~(size_t)255;
    return p;
  };
  _Float16* XCAT = (_Float16*)carve((size_t)RTOT * 512 * 2);  // x1..x4 concat (f16)
  _Float16* XYZH = (_Float16*)carve((size_t)RTOT * 32 * 2);   // xyz padded to 32
  float*    XX   = (float*)   carve((size_t)RTOT * 4);
  int*      IDX  = (int*)     carve((size_t)RTOT * TOPK * 4);
  _Float16* WT   = (_Float16*)carve((size_t)1024 * 512 * 2);  // largest weight (fuse)
  float*    S    = (float*)   carve(1024 * 4);
  float*    T    = (float*)   carve(1024 * 4);
  float*    G    = (float*)   carve(4 * 1024 * 4);
  float*    CB   = (float*)   carve(4 * 512 * 4);
  float*    UV   = (float*)   carve((size_t)RTOT * 512 * 4);  // 32MB, reused for f/h1/h2

  // ---- input conversion ----
  cvt_xyz<<<(RTOT * 32) / 256, 256, 0, stream>>>(xyz, XYZH, RTOT);

  // ---- EdgeConv layers ----
  const int cin[4]  = {32, 64, 64, 128};   // padded K-dims
  const int cout[4] = {64, 64, 128, 256};
  const int cinReal[4] = {3, 64, 64, 128};
  const int inOff[4]   = {0, 0, 64, 128};  // column offset in XCAT (layer 0 uses XYZH)
  const int outOff[4]  = {0, 64, 128, 256};

  for (int L = 0; L < 4; ++L) {
    const _Float16* feat = (L == 0) ? XYZH : (XCAT + inOff[L]);
    int ldf = (L == 0) ? 32 : 512;
    const float* w  = (const float*)d_in[ecBase[L] + 0];
    const float* bg = (const float*)d_in[ecBase[L] + 1];
    const float* bb = (const float*)d_in[ecBase[L] + 2];
    const float* bm = (const float*)d_in[ecBase[L] + 3];
    const float* bv = (const float*)d_in[ecBase[L] + 4];

    row_norms<<<RTOT / 4, 128, 0, stream>>>(feat, ldf, cin[L], XX, RTOT);
    knn_wmma<<<RTOT / 16, 32, 0, stream>>>(feat, ldf, cin[L], XX, IDX, NPTS);

    int wtElems = 2 * cout[L] * cin[L];
    prep_edge_w<<<(wtElems + 255) / 256, 256, 0, stream>>>(w, cout[L], cinReal[L], cin[L], WT);
    prep_bn<<<(cout[L] + 255) / 256, 256, 0, stream>>>(bg, bb, bm, bv, cout[L], S, T);

    int two_co = 2 * cout[L];
    launch_gemm(stream, feat, ldf, WT, cin[L], UV, nullptr, two_co,
                RTOT, two_co, cin[L], two_co,
                nullptr, nullptr, nullptr, 0, nullptr, 0);

    gather_max_bn<<<RTOT, cout[L], 0, stream>>>(UV, two_co, cout[L], IDX, S, T,
                                                XCAT, 512, outOff[L]);
  }

  // ---- fuse: [16384,512] x [512,1024] -> bn+lrelu -> f (f16, reuses UV) ----
  _Float16* F16 = (_Float16*)UV;
  prep_plain_w<<<(1024 * 512 + 255) / 256, 256, 0, stream>>>(fuse_w, 1024, 512, 512, 1024, WT);
  prep_bn<<<(1024 + 255) / 256, 256, 0, stream>>>((const float*)d_in[22], (const float*)d_in[23],
                                                  (const float*)d_in[24], (const float*)d_in[25],
                                                  1024, S, T);
  launch_gemm(stream, XCAT, 512, WT, 512, nullptr, F16, 1024,
              RTOT, 1024, 512, 1024, S, T, nullptr, 0, nullptr, 1);

  // ---- global max g, then per-batch bias c_b = W1b * g ----
  col_max<<<dim3(4, 1024 / 256), 256, 0, stream>>>(F16, 1024, NPTS, G);
  compute_cb<<<(4 * 512) / 256, 256, 0, stream>>>(h1_w, G, CB);

  // ---- h1: x_cat @ W1a^T + c_b -> bn+lrelu (f16, reuses UV[0:16MB)) ----
  _Float16* H1 = F16;
  prep_plain_w<<<(512 * 512 + 255) / 256, 256, 0, stream>>>(h1_w, 512, 1536, 512, 512, WT);
  prep_bn<<<(512 + 255) / 256, 256, 0, stream>>>((const float*)d_in[27], (const float*)d_in[28],
                                                 (const float*)d_in[29], (const float*)d_in[30],
                                                 512, S, T);
  launch_gemm(stream, XCAT, 512, WT, 512, nullptr, H1, 512,
              RTOT, 512, 512, 512, S, T, CB, 512, nullptr, 1);

  // ---- h2: [16384,512] x [512,256] -> bn+lrelu (f16, UV+16MB) ----
  _Float16* H2 = (_Float16*)((char*)UV + (size_t)RTOT * 512 * 2);
  prep_plain_w<<<(256 * 512 + 255) / 256, 256, 0, stream>>>(h2_w, 256, 512, 512, 256, WT);
  prep_bn<<<(256 + 255) / 256, 256, 0, stream>>>((const float*)d_in[32], (const float*)d_in[33],
                                                 (const float*)d_in[34], (const float*)d_in[35],
                                                 256, S, T);
  launch_gemm(stream, H1, 512, WT, 512, nullptr, H2, 256,
              RTOT, 256, 512, 256, S, T, nullptr, 0, nullptr, 1);

  // ---- h3: [16384,256] x [256,50(pad 64)] + bias -> logits f32 to d_out ----
  prep_plain_w<<<(64 * 256 + 255) / 256, 256, 0, stream>>>(h3_w, 50, 256, 256, 64, WT);
  launch_gemm(stream, H2, 256, WT, 256, (float*)d_out, nullptr, 50,
              RTOT, 64, 256, 50, nullptr, nullptr, nullptr, 0, h3_b, 0);
}
